// IntelligentHybridSSM_61065845015121
// MI455X (gfx1250) — matrix operand order repaired and checked by reading, not compiled, not run on hardware
//
#include <hip/hip_runtime.h>
#include <hip/hip_bf16.h>

// ---------------------------------------------------------------------------
// IntelligentHybridSSM for MI455X (gfx1250, wave32, WMMA)
//
//  prep : x -> bf16 ; weights -> bf16 transposed [N][K]
//  gemm : bf16 WMMA 16x16x32, f32 acc; 256x64 tile, wave = 32 rows x 64 cols
//         (2 A frags, 4 shared B frags -> 8 wmma / 12 ds_load_b128 per K-step;
//          each B frag feeds two back-to-back wmmas)
//  scan : one block per (b,h); 64x64 state in registers; y written as bf16
//  out  : y(bf16) @ Wo_t -> f32
// ---------------------------------------------------------------------------

#define DIM       1024
#define STATE_DIM 64
#define DT_RANK   32
#define HEAD_DIM  64
#define NUM_HEADS 8
#define BATCH     4
#define SEQ       4096
#define INNER     (NUM_HEADS * HEAD_DIM)    // 512
#define MTOT      (BATCH * SEQ)             // 16384
#define NSEL      (DT_RANK + 2 * STATE_DIM) // 160

typedef __attribute__((ext_vector_type(16))) __bf16        bf16x16;
typedef __attribute__((ext_vector_type(8)))  float         f32x8;
typedef __attribute__((ext_vector_type(4)))  unsigned int  u32x4;

#define TILE_M 256
#define TILE_N 64
#define TILE_K 32
#define LSTRIDE 40   // padded LDS row stride in bf16 (80 B: 16B-aligned, conflict-free)

union FragU { u32x4 u[2]; bf16x16 v; };

// ---------------------------------------------------------------------------
// C[M,N] = op( A[M,K](bf16, lda) * BT[N,K](bf16, ldb)^T )
// mode 0: identity   mode 1: sigmoid
// mode 2: exp(-scale[n] * softplus(acc + bias[n]))
// ---------------------------------------------------------------------------
__global__ __launch_bounds__(256) void wmma_gemm_kernel(
    const __bf16* __restrict__ A, int lda,
    const __bf16* __restrict__ BT, int ldb,
    float* __restrict__ C, int ldc,
    int M, int N, int K, int mode,
    const float* __restrict__ bias,
    const float* __restrict__ scale)
{
    __shared__ __align__(16) __bf16 lA[TILE_M * LSTRIDE];  // [m][k] padded
    __shared__ __align__(16) __bf16 lB[TILE_N * LSTRIDE];  // [n][k] padded

    const int t     = threadIdx.x;
    const int lane  = t & 31;
    const int wave  = t >> 5;
    const int half  = lane >> 4;   // 0/1
    const int l16   = lane & 15;
    const int tileM = blockIdx.y * TILE_M;
    const int tileN = blockIdx.x * TILE_N;
    const int rowB  = wave * 32;   // this wave: rows [rowB, rowB+32)

    f32x8 acc0[4], acc1[4];
#pragma unroll
    for (int j = 0; j < 4; ++j) {
        acc0[j] = (f32x8){0,0,0,0,0,0,0,0};
        acc1[j] = (f32x8){0,0,0,0,0,0,0,0};
    }

    for (int kt = 0; kt < K; kt += TILE_K) {
        // ---- stage A: 256 rows x 32 bf16 (64B/row); thread t copies row t ----
        {
            const u32x4* src = (const u32x4*)(A + (size_t)(tileM + t) * lda + kt);
            u32x4* dst = (u32x4*)&lA[t * LSTRIDE];
            dst[0] = src[0];
            dst[1] = src[1];
            dst[2] = src[2];
            dst[3] = src[3];
        }
        // ---- stage B: 64 rows (n) x 32 bf16 from BT[n][k]; thread: n=t>>2, chunk=t&3
        {
            const int n = t >> 2;
            const int c = t & 3;
            const int gn = tileN + n;
            u32x4 val;
            if (gn < N) val = ((const u32x4*)(BT + (size_t)gn * ldb + kt))[c];
            else        val = (u32x4){0u, 0u, 0u, 0u};
            ((u32x4*)&lB[n * LSTRIDE])[c] = val;
        }
        __syncthreads();

        // ---- two A fragments (rows rowB+l16 and rowB+16+l16) ----
        FragU a0, a1;
        {
            const u32x4* pa0 = (const u32x4*)&lA[(rowB + l16) * LSTRIDE];
            const u32x4* pa1 = (const u32x4*)&lA[(rowB + 16 + l16) * LSTRIDE];
            a0.u[0] = pa0[half];     a0.u[1] = pa0[2 + half];
            a1.u[0] = pa1[half];     a1.u[1] = pa1[2 + half];
        }
        // ---- 4 B fragments, each feeding two wmmas ----
#pragma unroll
        for (int j = 0; j < 4; ++j) {
            FragU bf;
            const u32x4* pb = (const u32x4*)&lB[(j * 16 + l16) * LSTRIDE];
            bf.u[0] = pb[half];
            bf.u[1] = pb[2 + half];
            acc0[j] = __builtin_amdgcn_wmma_f32_16x16x32_bf16(
                false, a0.v, false, bf.v, (short)0, acc0[j], false, false);
            acc1[j] = __builtin_amdgcn_wmma_f32_16x16x32_bf16(
                false, a1.v, false, bf.v, (short)0, acc1[j], false, false);
        }
        __syncthreads();
    }

    // ---- epilogue: acc{0,1}[j][r] -> (m, n = tileN + j*16 + l16) ----
#pragma unroll
    for (int j = 0; j < 4; ++j) {
        const int n = tileN + j * 16 + l16;
        if (n >= N) continue;
#pragma unroll
        for (int r8 = 0; r8 < 8; ++r8) {
#pragma unroll
            for (int blkm = 0; blkm < 2; ++blkm) {
                const int m = tileM + rowB + blkm * 16 + half * 8 + r8;
                float val = blkm ? acc1[j][r8] : acc0[j][r8];
                if (mode == 1) {
                    val = 1.0f / (1.0f + __expf(-val));
                } else if (mode == 2) {
                    const float d  = val + bias[n];
                    const float sp = (d > 20.0f) ? d : log1pf(__expf(d));
                    val = __expf(-scale[n] * sp);
                }
                C[(size_t)m * ldc + n] = val;
            }
        }
    }
}

// ---------------------------------------------------------------------------
// prep: f32 -> bf16 plain convert (row-major preserved)
// ---------------------------------------------------------------------------
__global__ __launch_bounds__(256) void cvt_bf16_kernel(
    const float* __restrict__ in, __bf16* __restrict__ out, long long n)
{
    const long long i = ((long long)blockIdx.x * blockDim.x + threadIdx.x) * 4;
    if (i + 3 >= n) {
        for (long long j = i; j < n; ++j) out[j] = (__bf16)in[j];
        return;
    }
    const float4 f = *(const float4*)(in + i);
    out[i]     = (__bf16)f.x;
    out[i + 1] = (__bf16)f.y;
    out[i + 2] = (__bf16)f.z;
    out[i + 3] = (__bf16)f.w;
}

// prep: out[n*K + k] = (bf16) in[k*N + n]   (transpose K x N -> N x K)
__global__ __launch_bounds__(256) void tr_bf16_kernel(
    const float* __restrict__ in, __bf16* __restrict__ out, int K, int N)
{
    const long long id = (long long)blockIdx.x * blockDim.x + threadIdx.x;
    if (id >= (long long)K * N) return;
    const int n = (int)(id / K);
    const int k = (int)(id % K);
    out[id] = (__bf16)in[(long long)k * N + n];
}

// ---------------------------------------------------------------------------
// bscale[m] = mean(proj[m, 32:160]) ; dt_bf[m, 0:32] = bf16(proj[m, 0:32])
// ---------------------------------------------------------------------------
__global__ __launch_bounds__(256) void bscale_kernel(
    const float* __restrict__ proj, float* __restrict__ bsc,
    __bf16* __restrict__ dtb, int M)
{
    const int m = blockIdx.x * blockDim.x + threadIdx.x;
    if (m >= M) return;
    const float* p = proj + (size_t)m * NSEL;
#pragma unroll
    for (int i = 0; i < DT_RANK; ++i) dtb[(size_t)m * DT_RANK + i] = (__bf16)p[i];
    float s = 0.0f;
#pragma unroll 4
    for (int i = 0; i < 2 * STATE_DIM; ++i) s += p[DT_RANK + i];
    bsc[m] = s * (1.0f / (2.0f * STATE_DIM));
}

// ---------------------------------------------------------------------------
// Sequential scan: block = one (b,h); 256 threads; v = t>>2, q = t&3
// ---------------------------------------------------------------------------
__global__ __launch_bounds__(256) void scan_kernel(
    const float* __restrict__ kb, const float* __restrict__ vb,
    const float* __restrict__ rb, const float* __restrict__ gb,
    const float* __restrict__ db, const float* __restrict__ bsc,
    const float* __restrict__ time_first,
    __bf16* __restrict__ yb, float* __restrict__ state_out)
{
    const int bh = blockIdx.x;            // 0..31
    const int b  = bh >> 3;
    const int h  = bh & (NUM_HEADS - 1);
    const int t  = threadIdx.x;
    const int v  = t >> 2;                // 0..63
    const int q  = t & 3;                 // 0..3
    const int k0 = q * 16;

    __shared__ float sk[HEAD_DIM];
    __shared__ float sd[HEAD_DIM];

    float st[16];
#pragma unroll
    for (int i = 0; i < 16; ++i) st[i] = 0.0f;

    const float tf = time_first[h * HEAD_DIM + v];
    const size_t base0 = (size_t)b * SEQ * INNER + h * HEAD_DIM;

    for (int l = 0; l < SEQ; ++l) {
        const size_t base = base0 + (size_t)l * INNER;
        if (t < 64)       sk[t]      = kb[base + t];
        else if (t < 128) sd[t - 64] = db[base + (t - 64)];
        __syncthreads();

        const float vt  = vb[base + v];
        const float bs  = bsc[b * SEQ + l];
        const float bkv = bs * vt;

        float partial = 0.0f;
#pragma unroll
        for (int i = 0; i < 16; ++i) {
            const float kv = sk[k0 + i];
            float s = st[i];
            s = sd[k0 + i] * s + bkv * kv;   // decay + rank-1 update
            st[i] = s;
            partial += s * kv;
        }
        partial += __shfl_xor(partial, 1, 32);
        partial += __shfl_xor(partial, 2, 32);

        if (q == 0) {
            const float wkv = partial + tf * sk[v] * vt;
            yb[base + v] = (__bf16)(rb[base + v] * wkv * gb[base + v]);
        }
        __syncthreads();
    }

#pragma unroll
    for (int i = 0; i < 16; ++i)
        state_out[((size_t)bh * HEAD_DIM + (k0 + i)) * HEAD_DIM + v] = st[i];
}

// ---------------------------------------------------------------------------
extern "C" void kernel_launch(void* const* d_in, const int* in_sizes, int n_in,
                              void* d_out, int out_size, void* d_ws, size_t ws_size,
                              hipStream_t stream)
{
    const float* x          = (const float*)d_in[0];
    const float* W_sel      = (const float*)d_in[1];
    const float* dt_W       = (const float*)d_in[2];
    const float* dt_b       = (const float*)d_in[3];
    const float* Wk         = (const float*)d_in[4];
    const float* Wv         = (const float*)d_in[5];
    const float* Wr         = (const float*)d_in[6];
    const float* Wg         = (const float*)d_in[7];
    const float* Wo         = (const float*)d_in[8];
    const float* base_decay = (const float*)d_in[9];
    const float* time_first = (const float*)d_in[10];

    float* out       = (float*)d_out;                    // (B,L,DIM)
    float* state_out = out + (size_t)MTOT * DIM;         // (B,H,64,64)

    // ---- workspace layout: f32 region first, then bf16 region ----
    float* ws     = (float*)d_ws;
    float* p_proj = ws;                                  // MTOT x 160
    float* p_k    = p_proj + (size_t)MTOT * NSEL;
    float* p_v    = p_k    + (size_t)MTOT * INNER;
    float* p_r    = p_v    + (size_t)MTOT * INNER;
    float* p_g    = p_r    + (size_t)MTOT * INNER;
    float* p_d    = p_g    + (size_t)MTOT * INNER;       // decay
    float* p_bs   = p_d    + (size_t)MTOT * INNER;       // bscale (MTOT)
    __bf16* x_bf  = (__bf16*)(p_bs + MTOT);              // MTOT x 1024
    __bf16* Wsel_t = x_bf   + (size_t)MTOT * DIM;        // 160 x 1024
    __bf16* Wk_t   = Wsel_t + (size_t)NSEL * DIM;        // 512 x 1024
    __bf16* Wv_t   = Wk_t   + (size_t)INNER * DIM;
    __bf16* Wr_t   = Wv_t   + (size_t)INNER * DIM;
    __bf16* Wg_t   = Wr_t   + (size_t)INNER * DIM;
    __bf16* dtW_t  = Wg_t   + (size_t)INNER * DIM;       // 512 x 32
    __bf16* Wo_t   = dtW_t  + (size_t)INNER * DT_RANK;   // 1024 x 512
    __bf16* dt_bf  = Wo_t   + (size_t)DIM * INNER;       // MTOT x 32
    __bf16* y_bf   = dt_bf  + (size_t)MTOT * DT_RANK;    // MTOT x 512

    const dim3 blk(256);

    // ---- prep: convert / transpose to bf16 ----
    {
        const long long nx = (long long)MTOT * DIM;
        cvt_bf16_kernel<<<dim3((unsigned)((nx / 4 + 255) / 256)), blk, 0, stream>>>(x, x_bf, nx);
        auto tr = [&](const float* src, __bf16* dst, int K, int N) {
            const long long e = (long long)K * N;
            tr_bf16_kernel<<<dim3((unsigned)((e + 255) / 256)), blk, 0, stream>>>(src, dst, K, N);
        };
        tr(W_sel, Wsel_t, DIM, NSEL);
        tr(Wk,    Wk_t,   DIM, INNER);
        tr(Wv,    Wv_t,   DIM, INNER);
        tr(Wr,    Wr_t,   DIM, INNER);
        tr(Wg,    Wg_t,   DIM, INNER);
        tr(dt_W,  dtW_t,  DT_RANK, INNER);
        tr(Wo,    Wo_t,   INNER, DIM);
    }

    const dim3 gSel((NSEL  + TILE_N - 1) / TILE_N, MTOT / TILE_M);
    const dim3 gInr((INNER + TILE_N - 1) / TILE_N, MTOT / TILE_M);
    const dim3 gOut((DIM   + TILE_N - 1) / TILE_N, MTOT / TILE_M);

    // ---- projections ----
    wmma_gemm_kernel<<<gSel, blk, 0, stream>>>(x_bf, DIM, Wsel_t, DIM, p_proj, NSEL,
                                               MTOT, NSEL, DIM, 0, nullptr, nullptr);
    wmma_gemm_kernel<<<gInr, blk, 0, stream>>>(x_bf, DIM, Wk_t, DIM, p_k, INNER,
                                               MTOT, INNER, DIM, 0, nullptr, nullptr);
    wmma_gemm_kernel<<<gInr, blk, 0, stream>>>(x_bf, DIM, Wv_t, DIM, p_v, INNER,
                                               MTOT, INNER, DIM, 0, nullptr, nullptr);
    wmma_gemm_kernel<<<gInr, blk, 0, stream>>>(x_bf, DIM, Wr_t, DIM, p_r, INNER,
                                               MTOT, INNER, DIM, 1, nullptr, nullptr);
    wmma_gemm_kernel<<<gInr, blk, 0, stream>>>(x_bf, DIM, Wg_t, DIM, p_g, INNER,
                                               MTOT, INNER, DIM, 1, nullptr, nullptr);

    // ---- bscale + compact bf16 dt_raw ----
    bscale_kernel<<<dim3((MTOT + 255) / 256), blk, 0, stream>>>(p_proj, p_bs, dt_bf, MTOT);

    // ---- decay = exp(-base_decay * softplus(dt_raw @ dt_W + dt_b)), K=32 ----
    wmma_gemm_kernel<<<gInr, blk, 0, stream>>>(dt_bf, DT_RANK, dtW_t, DT_RANK, p_d, INNER,
                                               MTOT, INNER, DT_RANK, 2, dt_b, base_decay);

    // ---- sequential scan ----
    scan_kernel<<<dim3(BATCH * NUM_HEADS), blk, 0, stream>>>(
        p_k, p_v, p_r, p_g, p_d, p_bs, time_first, y_bf, state_out);

    // ---- output projection ----
    wmma_gemm_kernel<<<gOut, blk, 0, stream>>>(y_bf, INNER, Wo_t, INNER, out, DIM,
                                               MTOT, DIM, INNER, 0, nullptr, nullptr);
}
